// TDLMAttention_11587821764671
// MI455X (gfx1250) — matrix operand order repaired and checked by reading, not compiled, not run on hardware
//
#include <hip/hip_runtime.h>
#include <hip/hip_bf16.h>
#include <stdint.h>

typedef __bf16 bf16_t;
typedef __attribute__((ext_vector_type(16))) __bf16 v16bf;
typedef __attribute__((ext_vector_type(8)))  __bf16 v8bf;
typedef __attribute__((ext_vector_type(8)))  float  v8f;
typedef __attribute__((ext_vector_type(4)))  unsigned int uint32x4_t;
typedef __attribute__((ext_vector_type(8)))  int int32x8_t;
typedef __attribute__((ext_vector_type(4)))  int int32x4_t;

static constexpr int B_  = 2;
static constexpr int S_  = 2048;
static constexpr int H_  = 2048;
static constexpr int NH_ = 16;
static constexpr int HD_ = 128;
static constexpr float SCALE_ = 0.08838834764831845f; // 1/sqrt(128)

// ---------------------------------------------------------------------------
// Tensor Data Mover: 2D tile (tile_x contiguous elems  x  tile_y rows) of a
// 2-byte-element tensor, global -> LDS, rows packed contiguously in LDS.
// D# layout per CDNA5 ISA ch.8 (group0: count/lds/global/type, group1:
// data_size, tensor dims, tile dims, dim0 stride).  Issued by one wave;
// completion via s_wait_tensorcnt + workgroup barrier.
// ---------------------------------------------------------------------------
static __device__ inline void tdm_load_2d(unsigned lds_off, const void* gsrc,
                                          unsigned tile_x, unsigned tile_y,
                                          unsigned tensor_x, unsigned tensor_y,
                                          unsigned stride_x_elems) {
  unsigned long long ga = (unsigned long long)(uintptr_t)gsrc;
  uint32x4_t g0;
  g0[0] = 1u;                                            // count=1, user D#
  g0[1] = lds_off;                                       // lds_addr (bytes)
  g0[2] = (unsigned)(ga & 0xffffffffu);                  // global_addr[31:0]
  g0[3] = (unsigned)((ga >> 32) & 0x01ffffffu) | (2u << 30); // addr[56:32] | type=2
  int32x8_t g1;
  g1[0] = (int)(1u << 16);                               // data_size=1 (2B), mask=0
  g1[1] = (int)((tensor_x & 0xffffu) << 16);             // tensor_dim0[15:0]
  g1[2] = (int)(((tensor_x >> 16) & 0xffffu) | ((tensor_y & 0xffffu) << 16));
  g1[3] = (int)(((tensor_y >> 16) & 0xffffu) | ((tile_x & 0xffffu) << 16));
  g1[4] = (int)(tile_y & 0xffffu);                       // tile_dim1, tile_dim2=0
  g1[5] = (int)stride_x_elems;                           // tensor_dim0_stride[31:0]
  g1[6] = 0;                                             // stride[47:32], dim1_stride
  g1[7] = 0;
  int32x4_t z4 = {0, 0, 0, 0};
#if defined(__clang_major__) && (__clang_major__ >= 23)
  int32x8_t z8 = {0, 0, 0, 0, 0, 0, 0, 0};
  __builtin_amdgcn_tensor_load_to_lds(g0, g1, z4, z4, z8, 0);
#else
  __builtin_amdgcn_tensor_load_to_lds(g0, g1, z4, z4, 0);
#endif
}

static __device__ inline unsigned lds_offset_of(const void* p) {
  // CDNA5 ISA 10.2: LDS aperture address truncates to addr[31:0] = LDS offset.
  return (unsigned)(uintptr_t)p;
}

// ---------------------------------------------------------------------------
// fp32 -> bf16 conversion
// ---------------------------------------------------------------------------
__global__ __launch_bounds__(256) void cvt_f32_bf16(const float* __restrict__ src,
                                                    bf16_t* __restrict__ dst, int n) {
  for (int i = blockIdx.x * blockDim.x + threadIdx.x; i < n;
       i += gridDim.x * blockDim.x)
    dst[i] = (bf16_t)src[i];
}

// ---------------------------------------------------------------------------
// RoPE on Q and K, layout [B,NH,S,HD] bf16.  Folds 1/sqrt(HD) into Q.
// ---------------------------------------------------------------------------
__global__ __launch_bounds__(256) void rope_kernel(bf16_t* __restrict__ q,
                                                   bf16_t* __restrict__ k) {
  int idx = blockIdx.x * blockDim.x + threadIdx.x;     // B*NH*S*64 threads
  int d  = idx & 63;
  int s  = (idx >> 6) & (S_ - 1);
  int bh = idx >> 17;
  float inv = __expf(-(float)d * (9.210340372f / 64.0f)); // 10000^(-d/64)
  float th = (float)s * inv;
  float c, sn;
  __sincosf(th, &c, &sn);
  size_t base = ((size_t)bh * S_ + s) * HD_;
  float q1 = (float)q[base + d], q2 = (float)q[base + d + 64];
  float k1 = (float)k[base + d], k2 = (float)k[base + d + 64];
  q[base + d]      = (bf16_t)((q1 * c - q2 * sn) * SCALE_);
  q[base + d + 64] = (bf16_t)((q2 * c + q1 * sn) * SCALE_);
  k[base + d]      = (bf16_t)(k1 * c - k2 * sn);
  k[base + d + 64] = (bf16_t)(k2 * c + k1 * sn);
}

// ---------------------------------------------------------------------------
// QKV projection: C = X(4096x2048) * W^T.  Block tile 128x64, 4 waves, each
// wave owns 32x64 (2 A-frags x 4 B-frags = 8 WMMA per K-step).  A/B tiles are
// staged by the Tensor Data Mover into double-buffered LDS; wave 0 issues the
// descriptors, everyone synchronizes via s_wait_tensorcnt + barrier.
// z=0 -> Q [B,NH,S,HD], z=1 -> K [B,NH,S,HD], z=2 -> V transposed [B,NH,HD,S].
// ---------------------------------------------------------------------------
__global__ __launch_bounds__(128) void gemm_qkv(const bf16_t* __restrict__ X,
                                                const bf16_t* __restrict__ Wq,
                                                const bf16_t* __restrict__ Wk,
                                                const bf16_t* __restrict__ Wv,
                                                bf16_t* __restrict__ Qh,
                                                bf16_t* __restrict__ Kh,
                                                bf16_t* __restrict__ Vt) {
  __shared__ alignas(32) bf16_t As[2][128][32];   // 16 KB
  __shared__ alignas(32) bf16_t Bs[2][64][32];    //  8 KB
  const int which = blockIdx.z;
  const bf16_t* W = (which == 0) ? Wq : (which == 1) ? Wk : Wv;
  const int tN = blockIdx.x * 64, tM = blockIdx.y * 128;
  const int lane = threadIdx.x & 31, wave = threadIdx.x >> 5;
  const int lm = lane & 15, lh = lane >> 4;

  v8f acc[2][4] = {};
  constexpr int NSTAGE = H_ / 32;

  if (wave == 0) {
    tdm_load_2d(lds_offset_of(&As[0][0][0]), X + (size_t)tM * H_,
                32, 128, H_, B_ * S_, H_);
    tdm_load_2d(lds_offset_of(&Bs[0][0][0]), W + (size_t)tN * H_,
                32, 64, H_, H_, H_);
    __builtin_amdgcn_s_wait_tensorcnt(0);
  }
  __syncthreads();

  for (int st = 0; st < NSTAGE; ++st) {
    const int cur = st & 1, nxt = cur ^ 1;
    if ((st + 1 < NSTAGE) && wave == 0) {      // overlap next-stage DMA w/ compute
      int k0 = (st + 1) * 32;
      tdm_load_2d(lds_offset_of(&As[nxt][0][0]), X + (size_t)tM * H_ + k0,
                  32, 128, H_, B_ * S_, H_);
      tdm_load_2d(lds_offset_of(&Bs[nxt][0][0]), W + (size_t)tN * H_ + k0,
                  32, 64, H_, H_, H_);
    }

    // B fragments (shared across both M-subtiles of this wave)
    v16bf bfr[4];
    #pragma unroll
    for (int t = 0; t < 4; ++t)
      bfr[t] = *(const v16bf*)&Bs[cur][16 * t + lm][16 * lh];

    #pragma unroll
    for (int sub = 0; sub < 2; ++sub) {
      union { v16bf v; v8bf h[2]; } au;
      const bf16_t* arow = &As[cur][32 * wave + 16 * sub + lm][0];
      au.h[0] = *(const v8bf*)(arow + 16 * lh);
      au.h[1] = *(const v8bf*)(arow + 16 * lh + 8);
      v16bf af = au.v;
      #pragma unroll
      for (int t = 0; t < 4; ++t)
        acc[sub][t] = __builtin_amdgcn_wmma_f32_16x16x32_bf16(
            false, af, false, bfr[t], (short)0, acc[sub][t], false, false);
    }

    if (wave == 0) __builtin_amdgcn_s_wait_tensorcnt(0);
    __syncthreads();
  }

  bf16_t* dstQK = (which == 0) ? Qh : Kh;
  #pragma unroll
  for (int sub = 0; sub < 2; ++sub)
    #pragma unroll
    for (int t = 0; t < 4; ++t)
      #pragma unroll
      for (int r = 0; r < 8; ++r) {
        int m = tM + 32 * wave + 16 * sub + r + 8 * lh;
        int n = tN + 16 * t + lm;
        int b = m >> 11, s = m & (S_ - 1);
        int hh = n >> 7, d = n & (HD_ - 1);
        bf16_t v = (bf16_t)acc[sub][t][r];
        if (which == 2)
          Vt[((size_t)(b * NH_ + hh) * HD_ + d) * S_ + s] = v;
        else
          dstQK[((size_t)(b * NH_ + hh) * S_ + s) * HD_ + d] = v;
      }
}

// ---------------------------------------------------------------------------
// Flash attention.  8 waves/block, each wave owns a 16-row Q tile.
// Streams 32-key tiles with online softmax; O accumulated in f32 (8x v8f).
// ---------------------------------------------------------------------------
__global__ __launch_bounds__(256) void flash_kernel(const bf16_t* __restrict__ Qh,
                                                    const bf16_t* __restrict__ Kh,
                                                    const bf16_t* __restrict__ Vt,
                                                    const float* __restrict__ mask,
                                                    bf16_t* __restrict__ O) {
  __shared__ alignas(32) bf16_t Ps[8][16][32];
  const int lane = threadIdx.x & 31, wave = threadIdx.x >> 5;
  const int lm = lane & 15, lh = lane >> 4;
  const int bh = blockIdx.y;
  const int b = bh >> 4, hh = bh & 15;
  const int q0 = blockIdx.x * 128 + wave * 16;

  const bf16_t* Qp = Qh + (size_t)bh * S_ * HD_;
  const bf16_t* Kp = Kh + (size_t)bh * S_ * HD_;
  const bf16_t* Vp = Vt + (size_t)bh * HD_ * S_;

  v16bf qf[4];
  const bf16_t* qrow = Qp + (size_t)(q0 + lm) * HD_;
  #pragma unroll
  for (int kk = 0; kk < 4; ++kk) {
    union { v16bf v; v8bf h[2]; } u;
    int dbase = kk * 32 + 16 * lh;
    u.h[0] = *(const v8bf*)(qrow + dbase);
    u.h[1] = *(const v8bf*)(qrow + dbase + 8);
    qf[kk] = u.v;
  }

  v8f acc[8] = {};
  float mrow[8], lrow[8];
  #pragma unroll
  for (int r = 0; r < 8; ++r) { mrow[r] = -1e30f; lrow[r] = 0.0f; }

  const int nkt = (q0 + 16 + 31) >> 5;   // causal: keys <= q only
  for (int kt = 0; kt < nkt; ++kt) {
    const int k0 = kt * 32;

    v8f s0 = {}, s1 = {};
    #pragma unroll
    for (int kk = 0; kk < 4; ++kk) {
      int dbase = kk * 32 + 16 * lh;
      v16bf b0 = *(const v16bf*)(Kp + (size_t)(k0 + lm) * HD_ + dbase);
      v16bf b1 = *(const v16bf*)(Kp + (size_t)(k0 + 16 + lm) * HD_ + dbase);
      s0 = __builtin_amdgcn_wmma_f32_16x16x32_bf16(false, qf[kk], false, b0,
                                                   (short)0, s0, false, false);
      s1 = __builtin_amdgcn_wmma_f32_16x16x32_bf16(false, qf[kk], false, b1,
                                                   (short)0, s1, false, false);
    }

    float am0 = (1.0f - mask[b * S_ + k0 + lm]) * -10000.0f;
    float am1 = (1.0f - mask[b * S_ + k0 + 16 + lm]) * -10000.0f;

    #pragma unroll
    for (int r = 0; r < 8; ++r) {
      int qg = q0 + r + 8 * lh;
      float e0 = s0[r] + am0 + ((k0 + lm) <= qg ? 0.0f : -1e30f);
      float e1 = s1[r] + am1 + ((k0 + 16 + lm) <= qg ? 0.0f : -1e30f);
      float mx = fmaxf(e0, e1);
      #pragma unroll
      for (int off = 1; off < 16; off <<= 1)
        mx = fmaxf(mx, __shfl_xor(mx, off, 32));
      float mnew = fmaxf(mrow[r], mx);
      float sc = __expf(mrow[r] - mnew);
      float p0 = __expf(e0 - mnew);
      float p1 = __expf(e1 - mnew);
      float rs = p0 + p1;
      #pragma unroll
      for (int off = 1; off < 16; off <<= 1)
        rs += __shfl_xor(rs, off, 32);
      lrow[r] = lrow[r] * sc + rs;
      mrow[r] = mnew;
      #pragma unroll
      for (int t = 0; t < 8; ++t) acc[t][r] *= sc;
      int m = r + 8 * lh;
      Ps[wave][m][lm]      = (bf16_t)p0;
      Ps[wave][m][16 + lm] = (bf16_t)p1;
    }
    asm volatile("s_wait_dscnt 0x0" ::: "memory");

    union { v16bf v; v8bf h[2]; } pu;
    const bf16_t* prow = &Ps[wave][lm][0];
    pu.h[0] = *(const v8bf*)(prow + 16 * lh);
    pu.h[1] = *(const v8bf*)(prow + 16 * lh + 8);
    v16bf pf = pu.v;

    #pragma unroll
    for (int t = 0; t < 8; ++t) {
      v16bf bv = *(const v16bf*)(Vp + (size_t)(t * 16 + lm) * S_ + k0 + 16 * lh);
      acc[t] = __builtin_amdgcn_wmma_f32_16x16x32_bf16(false, pf, false, bv,
                                                       (short)0, acc[t], false, false);
    }
  }

  #pragma unroll
  for (int t = 0; t < 8; ++t) {
    #pragma unroll
    for (int r = 0; r < 8; ++r) {
      int m = r + 8 * lh, d = t * 16 + lm;
      O[((size_t)(b * S_ + q0 + m)) * H_ + hh * HD_ + d] =
          (bf16_t)(acc[t][r] / lrow[r]);
    }
  }
}

// ---------------------------------------------------------------------------
// Output projection: out(f32) = O_bf16 * Wo^T  (same TDM-staged structure)
// ---------------------------------------------------------------------------
__global__ __launch_bounds__(128) void gemm_out(const bf16_t* __restrict__ X,
                                                const bf16_t* __restrict__ W,
                                                float* __restrict__ out) {
  __shared__ alignas(32) bf16_t As[2][128][32];
  __shared__ alignas(32) bf16_t Bs[2][64][32];
  const int tN = blockIdx.x * 64, tM = blockIdx.y * 128;
  const int lane = threadIdx.x & 31, wave = threadIdx.x >> 5;
  const int lm = lane & 15, lh = lane >> 4;

  v8f acc[2][4] = {};
  constexpr int NSTAGE = H_ / 32;

  if (wave == 0) {
    tdm_load_2d(lds_offset_of(&As[0][0][0]), X + (size_t)tM * H_,
                32, 128, H_, B_ * S_, H_);
    tdm_load_2d(lds_offset_of(&Bs[0][0][0]), W + (size_t)tN * H_,
                32, 64, H_, H_, H_);
    __builtin_amdgcn_s_wait_tensorcnt(0);
  }
  __syncthreads();

  for (int st = 0; st < NSTAGE; ++st) {
    const int cur = st & 1, nxt = cur ^ 1;
    if ((st + 1 < NSTAGE) && wave == 0) {
      int k0 = (st + 1) * 32;
      tdm_load_2d(lds_offset_of(&As[nxt][0][0]), X + (size_t)tM * H_ + k0,
                  32, 128, H_, B_ * S_, H_);
      tdm_load_2d(lds_offset_of(&Bs[nxt][0][0]), W + (size_t)tN * H_ + k0,
                  32, 64, H_, H_, H_);
    }

    v16bf bfr[4];
    #pragma unroll
    for (int t = 0; t < 4; ++t)
      bfr[t] = *(const v16bf*)&Bs[cur][16 * t + lm][16 * lh];

    #pragma unroll
    for (int sub = 0; sub < 2; ++sub) {
      union { v16bf v; v8bf h[2]; } au;
      const bf16_t* arow = &As[cur][32 * wave + 16 * sub + lm][0];
      au.h[0] = *(const v8bf*)(arow + 16 * lh);
      au.h[1] = *(const v8bf*)(arow + 16 * lh + 8);
      v16bf af = au.v;
      #pragma unroll
      for (int t = 0; t < 4; ++t)
        acc[sub][t] = __builtin_amdgcn_wmma_f32_16x16x32_bf16(
            false, af, false, bfr[t], (short)0, acc[sub][t], false, false);
    }

    if (wave == 0) __builtin_amdgcn_s_wait_tensorcnt(0);
    __syncthreads();
  }

  #pragma unroll
  for (int sub = 0; sub < 2; ++sub)
    #pragma unroll
    for (int t = 0; t < 4; ++t)
      #pragma unroll
      for (int r = 0; r < 8; ++r) {
        int m = tM + 32 * wave + 16 * sub + r + 8 * lh;
        int n = tN + 16 * t + lm;
        out[(size_t)m * H_ + n] = acc[sub][t][r];
      }
}

// ---------------------------------------------------------------------------
extern "C" void kernel_launch(void* const* d_in, const int* in_sizes, int n_in,
                              void* d_out, int out_size, void* d_ws, size_t ws_size,
                              hipStream_t stream) {
  const float* hx    = (const float*)d_in[0];  // [B,S,H]
  const float* amask = (const float*)d_in[1];  // [B,S]
  const float* wq    = (const float*)d_in[2];
  const float* wk    = (const float*)d_in[3];
  const float* wv    = (const float*)d_in[4];
  const float* wo    = (const float*)d_in[5];
  float* out = (float*)d_out;

  const size_t nX = (size_t)B_ * S_ * H_;     // 8,388,608
  const size_t nW = (size_t)H_ * H_;          // 4,194,304

  char* p = (char*)d_ws;
  bf16_t* Xb  = (bf16_t*)p; p += nX * 2;
  bf16_t* Wqb = (bf16_t*)p; p += nW * 2;
  bf16_t* Wkb = (bf16_t*)p; p += nW * 2;
  bf16_t* Wvb = (bf16_t*)p; p += nW * 2;
  bf16_t* Wob = (bf16_t*)p; p += nW * 2;
  bf16_t* Qh  = (bf16_t*)p; p += nX * 2;
  bf16_t* Kh  = (bf16_t*)p; p += nX * 2;
  bf16_t* Vt  = (bf16_t*)p; p += nX * 2;
  bf16_t* Ob  = (bf16_t*)p; p += nX * 2;      // total ~117 MB

  cvt_f32_bf16<<<2048, 256, 0, stream>>>(hx, Xb, (int)nX);
  cvt_f32_bf16<<<2048, 256, 0, stream>>>(wq, Wqb, (int)nW);
  cvt_f32_bf16<<<2048, 256, 0, stream>>>(wk, Wkb, (int)nW);
  cvt_f32_bf16<<<2048, 256, 0, stream>>>(wv, Wvb, (int)nW);
  cvt_f32_bf16<<<2048, 256, 0, stream>>>(wo, Wob, (int)nW);

  gemm_qkv<<<dim3(H_ / 64, (B_ * S_) / 128, 3), 128, 0, stream>>>(
      Xb, Wqb, Wkb, Wvb, Qh, Kh, Vt);

  rope_kernel<<<(B_ * NH_ * S_ * 64) / 256, 256, 0, stream>>>(Qh, Kh);

  flash_kernel<<<dim3(S_ / 128, B_ * NH_), 256, 0, stream>>>(
      Qh, Kh, Vt, amask, Ob);

  gemm_out<<<dim3(H_ / 64, (B_ * S_) / 128), 128, 0, stream>>>(Ob, Wob, out);
}
